// BatchShawMultigraphAttention_67774583931430
// MI455X (gfx1250) — matrix operand order repaired and checked by reading, not compiled, not run on hardware
//
#include <hip/hip_runtime.h>
#include <hip/hip_bf16.h>

// ---------------------------------------------------------------------------
// BatchShawMultigraphAttention, MI455X (gfx1250, wave32, WMMA).
//
// Math notes vs reference:
//  * attn_biases adds a row-constant to pre-softmax logits -> cancels. Unused.
//  * softmax rows sum to 1 -> value-bias term == biases[e,h,:] broadcast add.
// Precision: fp16 operands, f32 accumulation (v_wmma_f32_16x16x32_f16).
// ---------------------------------------------------------------------------

typedef __attribute__((ext_vector_type(16))) _Float16 v16h;
typedef __attribute__((ext_vector_type(8)))  _Float16 v8h;
typedef __attribute__((ext_vector_type(8)))  float    v8f;

#define B_    4
#define E_    4
#define H_    4
#define N_    1024
#define F_IN  64
#define F_OUT 32
#define OUTC  (E_ * H_ * F_OUT)   // 512

__device__ __forceinline__ v8f wmma_f16(v16h a, v16h b, v8f c) {
  return __builtin_amdgcn_wmma_f32_16x16x32_f16(false, a, false, b, (short)0, c,
                                                false, false);
}

__device__ __forceinline__ v16h ld16(const _Float16* p0, const _Float16* p1) {
  v8h lo = *(const v8h*)p0;
  v8h hi = *(const v8h*)p1;
  v16h r;
#pragma unroll
  for (int i = 0; i < 8; ++i) { r[i] = lo[i]; r[8 + i] = hi[i]; }
  return r;
}

// Butterfly reductions over the 16-lane half-groups (masks < 16 never cross
// the half boundary). Emits ds_bpermute_b32.
__device__ __forceinline__ float red16_max(float v) {
#pragma unroll
  for (int m = 8; m >= 1; m >>= 1) {
    int idx = (((int)(threadIdx.x & 31) ^ m) << 2);
    v = fmaxf(v, __int_as_float(__builtin_amdgcn_ds_bpermute(idx, __float_as_int(v))));
  }
  return v;
}
__device__ __forceinline__ float red16_sum(float v) {
#pragma unroll
  for (int m = 8; m >= 1; m >>= 1) {
    int idx = (((int)(threadIdx.x & 31) ^ m) << 2);
    v += __int_as_float(__builtin_amdgcn_ds_bpermute(idx, __float_as_int(v)));
  }
  return v;
}

// ---------------------------------------------------------------------------
// Kernel 0: weights [3][H][F][F_] f32 -> fp16 transposed [3][H][F_][F]
// ---------------------------------------------------------------------------
__global__ void prep_weights(const float* __restrict__ wq,
                             const float* __restrict__ wk,
                             const float* __restrict__ wv,
                             _Float16* __restrict__ wt) {
  int idx = blockIdx.x * blockDim.x + threadIdx.x;
  if (idx >= 3 * H_ * F_OUT * F_IN) return;
  int k = idx & (F_IN - 1);
  int n = (idx >> 6) & (F_OUT - 1);
  int h = (idx >> 11) & (H_ - 1);
  int m = idx >> 13;
  const float* src = (m == 0) ? wq : (m == 1) ? wk : wv;
  wt[idx] = (_Float16)src[(h * F_IN + k) * F_OUT + n];
}

// ---------------------------------------------------------------------------
// Kernel 1: Q/K/V projections, one wave = 16 tokens x 32 features.
//   z = 0: Q (attn_kernel_self)   -> qb[bh][tok][32]
//   z = 1: K (attn_kernel_neighs) -> kb[bh][tok][32]
//   z = 2: V (kernel)             -> vt[bh][feat][1024]  (transposed)
// ---------------------------------------------------------------------------
__global__ __launch_bounds__(128) void qkv_proj(const float* __restrict__ X,
                                                const _Float16* __restrict__ wt,
                                                _Float16* __restrict__ qb,
                                                _Float16* __restrict__ kb,
                                                _Float16* __restrict__ vt) {
  const int lane = threadIdx.x & 31;
  const int wid  = threadIdx.x >> 5;
  const int l16  = lane & 15;
  const int hl   = lane >> 4;
  const int bh   = blockIdx.y;
  const int b    = bh >> 2;
  const int m    = blockIdx.z;
  const int r0   = (blockIdx.x * 4 + wid) * 16;

  // B-fragments of W^T: lane = output feature n, table-order K (8h+s, 16+8h+s-8).
  const _Float16* wbase = wt + (size_t)((m * H_ + (bh & 3)) * F_OUT) * F_IN;
  v16h bw[2][2];
#pragma unroll
  for (int t = 0; t < 2; ++t)
#pragma unroll
    for (int kk = 0; kk < 2; ++kk) {
      const _Float16* p = wbase + (t * 16 + l16) * F_IN + kk * 32 + hl * 8;
      bw[t][kk] = ld16(p, p + 16);
    }

  // A-fragments of X (convert f32 -> f16), lane = token row.
  const float* xr = X + (size_t)(b * N_ + r0 + l16) * F_IN;
  v16h ax[2];
#pragma unroll
  for (int kk = 0; kk < 2; ++kk) {
    v16h a;
#pragma unroll
    for (int i = 0; i < 8; ++i) {
      a[i]     = (_Float16)xr[kk * 32 + hl * 8 + i];
      a[8 + i] = (_Float16)xr[kk * 32 + 16 + hl * 8 + i];
    }
    ax[kk] = a;
  }

  v8f acc0 = {0, 0, 0, 0, 0, 0, 0, 0};
  v8f acc1 = {0, 0, 0, 0, 0, 0, 0, 0};
  acc0 = wmma_f16(ax[0], bw[0][0], acc0);
  acc0 = wmma_f16(ax[1], bw[0][1], acc0);
  acc1 = wmma_f16(ax[0], bw[1][0], acc1);
  acc1 = wmma_f16(ax[1], bw[1][1], acc1);

  if (m < 2) {
    _Float16* dst = (m == 0 ? qb : kb) + (size_t)bh * N_ * F_OUT;
#pragma unroll
    for (int g = 0; g < 8; ++g) {
      int row = r0 + g + 8 * hl;
      dst[row * F_OUT + l16]      = (_Float16)acc0[g];
      dst[row * F_OUT + 16 + l16] = (_Float16)acc1[g];
    }
  } else {
    _Float16* dst = vt + (size_t)bh * F_OUT * N_;
#pragma unroll
    for (int g = 0; g < 8; ++g) {
      int tok = r0 + g + 8 * hl;
      dst[l16 * N_ + tok]        = (_Float16)acc0[g];
      dst[(16 + l16) * N_ + tok] = (_Float16)acc1[g];
    }
  }
}

// ---------------------------------------------------------------------------
// Kernel 2: fused masked flash attention. Block = 4 waves, wave = edge type e.
// One block handles (b, h, 16-query tile); loop over 32-key chunks.
// ---------------------------------------------------------------------------
__global__ __launch_bounds__(128) void mga_attn(const float* __restrict__ A,
                                                const float* __restrict__ biases,
                                                const _Float16* __restrict__ qb,
                                                const _Float16* __restrict__ kb,
                                                const _Float16* __restrict__ vt,
                                                float* __restrict__ out) {
  __shared__ _Float16 Plds[4][16 * 32];   // per-wave P-transpose slab (1 KB each)

  const int lane = threadIdx.x & 31;
  const int e    = threadIdx.x >> 5;      // edge type per wave
  const int l16  = lane & 15;
  const int hl   = lane >> 4;
  const int bh   = blockIdx.y;
  const int b    = bh >> 2;
  const int h    = bh & 3;
  const int i0   = blockIdx.x * 16;
  const float scale = 0.17677669529663687f;  // 1/sqrt(F_)

  // Q A-fragment, loaded once.
  const _Float16* qrow = qb + (size_t)(bh * N_ + i0 + l16) * F_OUT;
  const v16h aq = ld16(qrow + hl * 16, qrow + hl * 16 + 8);

  v8f acc0 = {0, 0, 0, 0, 0, 0, 0, 0};
  v8f acc1 = {0, 0, 0, 0, 0, 0, 0, 0};
  float mrow[8], lrow[8];
#pragma unroll
  for (int g = 0; g < 8; ++g) { mrow[g] = -3.0e38f; lrow[g] = 0.0f; }

  const float* Abase = A + ((size_t)(b * E_ + e) * N_ + i0) * N_;

  for (int j0 = 0; j0 < N_; j0 += 32) {
    // K B-fragments: lane = key, contiguous features per half.
    const _Float16* kbase = kb + (size_t)(bh * N_ + j0) * F_OUT;
    const _Float16* kp0 = kbase + l16 * F_OUT + hl * 16;
    const _Float16* kp1 = kbase + (16 + l16) * F_OUT + hl * 16;
    v8f s0 = {0, 0, 0, 0, 0, 0, 0, 0};
    v8f s1 = {0, 0, 0, 0, 0, 0, 0, 0};
    s0 = wmma_f16(aq, ld16(kp0, kp0 + 8), s0);
    s1 = wmma_f16(aq, ld16(kp1, kp1 + 8), s1);

    // V B-fragments from transposed V: lane = feature, contiguous keys.
    const _Float16* vbase = vt + (size_t)bh * F_OUT * N_ + j0 + hl * 16;
    const _Float16* vp0 = vbase + (size_t)l16 * N_;
    const _Float16* vp1 = vbase + (size_t)(16 + l16) * N_;
    v16h bv0 = ld16(vp0, vp0 + 8);
    v16h bv1 = ld16(vp1, vp1 + 8);

    if (j0 + 32 < N_)  // prefetch next chunk's mask rows -> global_prefetch_b8
      __builtin_prefetch(Abase + (size_t)(8 * hl) * N_ + j0 + 32 + l16, 0, 1);

    // Online softmax per row g (+8*hl); column j = l16 (+16 for subtile 1).
    float p0[8], p1[8];
#pragma unroll
    for (int g = 0; g < 8; ++g) {
      const int r = g + 8 * hl;
      float a0 = Abase[(size_t)r * N_ + j0 + l16];
      float a1 = Abase[(size_t)r * N_ + j0 + 16 + l16];
      float sv0 = s0[g] * scale + (a0 != 0.0f ? 0.0f : -1.0e10f);
      float sv1 = s1[g] * scale + (a1 != 0.0f ? 0.0f : -1.0e10f);
      float mn  = fmaxf(mrow[g], red16_max(fmaxf(sv0, sv1)));
      float e0  = __expf(sv0 - mn);
      float e1  = __expf(sv1 - mn);
      float rs  = red16_sum(e0 + e1);
      float fac = __expf(mrow[g] - mn);
      lrow[g] = lrow[g] * fac + rs;
      mrow[g] = mn;
      acc0[g] *= fac;
      acc1[g] *= fac;
      p0[g] = e0;
      p1[g] = e1;
    }

    // Transpose P (C/D layout -> A-fragment layout) through per-wave LDS.
    // Same-wave DS ops are processed in order; wave_barrier pins scheduling.
    _Float16* P = Plds[e];
#pragma unroll
    for (int g = 0; g < 8; ++g) {
      const int r = g + 8 * hl;
      P[r * 32 + l16]      = (_Float16)p0[g];
      P[r * 32 + 16 + l16] = (_Float16)p1[g];
    }
    __builtin_amdgcn_wave_barrier();
    const _Float16* pr = P + l16 * 32 + hl * 16;
    v16h ap = ld16(pr, pr + 8);
    __builtin_amdgcn_wave_barrier();

    acc0 = wmma_f16(ap, bv0, acc0);
    acc1 = wmma_f16(ap, bv1, acc1);
  }

  // Epilogue: normalize, add per-(e,h) value bias (softmax row sum == 1), ReLU.
  const float bias0 = biases[(e * H_ + h) * F_OUT + l16];
  const float bias1 = biases[(e * H_ + h) * F_OUT + 16 + l16];
  float* obase = out + ((size_t)b * N_ + i0) * OUTC + (e * H_ + h) * F_OUT;
#pragma unroll
  for (int g = 0; g < 8; ++g) {
    const int r = g + 8 * hl;
    float inv = 1.0f / lrow[g];
    obase[(size_t)r * OUTC + l16]      = fmaxf(acc0[g] * inv + bias0, 0.0f);
    obase[(size_t)r * OUTC + 16 + l16] = fmaxf(acc1[g] * inv + bias1, 0.0f);
  }
}

// ---------------------------------------------------------------------------
extern "C" void kernel_launch(void* const* d_in, const int* in_sizes, int n_in,
                              void* d_out, int out_size, void* d_ws, size_t ws_size,
                              hipStream_t stream) {
  const float* X    = (const float*)d_in[0];
  const float* A    = (const float*)d_in[1];
  const float* Wv   = (const float*)d_in[2];  // kernel       -> V
  const float* bias = (const float*)d_in[3];  // biases
  const float* Wq   = (const float*)d_in[4];  // attn_kernel_self   -> Q
  const float* Wk   = (const float*)d_in[5];  // attn_kernel_neighs -> K
  // d_in[6] attn_biases: row-constant pre-softmax -> cancels, unused.
  float* out = (float*)d_out;

  char* ws = (char*)d_ws;
  const size_t WT_BYTES  = (size_t)3 * H_ * F_OUT * F_IN * sizeof(_Float16); // 48 KB
  const size_t MAT_BYTES = (size_t)B_ * H_ * N_ * F_OUT * sizeof(_Float16);  // 1 MB
  _Float16* wt = (_Float16*)ws;
  _Float16* qb = (_Float16*)(ws + 49152);
  _Float16* kb = (_Float16*)(ws + 49152 + MAT_BYTES);
  _Float16* vt = (_Float16*)(ws + 49152 + 2 * MAT_BYTES);
  (void)WT_BYTES; (void)in_sizes; (void)n_in; (void)out_size; (void)ws_size;

  hipLaunchKernelGGL(prep_weights, dim3(96), dim3(256), 0, stream, Wq, Wk, Wv, wt);
  hipLaunchKernelGGL(qkv_proj, dim3(16, 16, 3), dim3(128), 0, stream,
                     X, wt, qb, kb, vt);
  hipLaunchKernelGGL(mga_attn, dim3(64, 16), dim3(128), 0, stream,
                     A, bias, qb, kb, vt, out);
}
